// MoELinear_64854006170287
// MI455X (gfx1250) — compile-verified, matrix operand order checked by reading, not run on hardware
//
#include <hip/hip_runtime.h>

typedef __attribute__((ext_vector_type(16))) __bf16        v16bf;
typedef __attribute__((ext_vector_type(8)))  float         v8f;
typedef __attribute__((ext_vector_type(8)))  unsigned int  v8u;
typedef __attribute__((ext_vector_type(4)))  unsigned int  v4u;
typedef unsigned int __attribute__((may_alias)) uint_alias;
typedef v4u __attribute__((may_alias))          v4u_alias;

#define NEXP 8
#define MTOK 8192
#define KIN  2048
#define NOUT 8192
#define BM 128
#define BN 128
#define BK 32
#define SA 40                 // padded K-stride (bf16 elems): 80B rows -> 16B aligned, 20-dword bank step
#define KBLOCKS (KIN / BK)

__device__ __forceinline__ unsigned int pack2(__bf16 a, __bf16 b) {
  const unsigned short ua = __builtin_bit_cast(unsigned short, a);
  const unsigned short ub = __builtin_bit_cast(unsigned short, b);
  return (unsigned int)ua | ((unsigned int)ub << 16);
}

__device__ __forceinline__ void bsplit(float v, __bf16& h, __bf16& l) {
  h = (__bf16)v;
  l = (__bf16)(v - (float)h);
}

// One 16x32 bf16 WMMA operand fragment: two aligned b128 LDS loads.
// VGPRs 0..3 hold K pairs {0,2,4,6}+8*half, VGPRs 4..7 hold {16,18,20,22}+8*half.
__device__ __forceinline__ v16bf load_frag(const __bf16* base, int half) {
  const v4u lo = *(const v4u_alias*)(base + (half << 3));
  const v4u hi = *(const v4u_alias*)(base + 16 + (half << 3));
  v8u t;
  t[0] = lo[0]; t[1] = lo[1]; t[2] = lo[2]; t[3] = lo[3];
  t[4] = hi[0]; t[5] = hi[1]; t[6] = hi[2]; t[7] = hi[3];
  return __builtin_bit_cast(v16bf, t);
}

__global__ __launch_bounds__(256)
void moe_grouped_gemm_bf16x3(const float* __restrict__ X,
                             const float* __restrict__ W,
                             const int*   __restrict__ GS,
                             const float* __restrict__ bias,
                             float*       __restrict__ out) {
  // Double-buffered hi/lo tiles: A as [m][k], B as [n][k] (transposed). 80 KB total.
  __shared__ __align__(16) __bf16 AHs[2][BM * SA], ALs[2][BM * SA];
  __shared__ __align__(16) __bf16 BHs[2][BN * SA], BLs[2][BN * SA];

  const int n0 = blockIdx.x * BN;
  const int m0 = blockIdx.y * BM;

  // ---- expert lookup for this row tile (tokens contiguous per expert) ----
  int e = 0, s = 0;
#pragma unroll
  for (int i = 0; i < NEXP; ++i) { s += GS[i]; if (m0 >= s) e = i + 1; }
  const float* Wp = W + (size_t)e * KIN * NOUT;

  const int t = threadIdx.x;
  // A producer: 8 threads/row, one float4 each; rows ar+32i
  const int ar = t >> 3, ac = (t & 7) * 4;
  // B producer: n-quads across the wave (coalesced), k-pairs per thread
  const int bn4 = (t & 31) * 4;          // n offset (quad)
  const int bkp = (t >> 5);              // k-pair index 0..7 -> k = 2*bkp (+16i)

  float4 aReg[4], bReg[4];

  auto loadRegs = [&](int kb) {
    const float* ap = X + (size_t)(m0 + ar) * KIN + kb * BK + ac;
#pragma unroll
    for (int i = 0; i < 4; ++i)
      aReg[i] = *(const float4*)(ap + (size_t)(32 * i) * KIN);
    const float* bp = Wp + (size_t)(kb * BK + 2 * bkp) * NOUT + n0 + bn4;
    bReg[0] = *(const float4*)(bp);
    bReg[1] = *(const float4*)(bp + NOUT);
    bReg[2] = *(const float4*)(bp + (size_t)16 * NOUT);
    bReg[3] = *(const float4*)(bp + (size_t)17 * NOUT);
  };

  auto stageLDS = [&](int buf) {
    // A: pack 4 consecutive k as two dwords per row (hi and lo tiles)
#pragma unroll
    for (int i = 0; i < 4; ++i) {
      const int row = ar + 32 * i;
      const float* f = (const float*)&aReg[i];
      __bf16 h[4], l[4];
#pragma unroll
      for (int j = 0; j < 4; ++j) bsplit(f[j], h[j], l[j]);
      uint_alias* pH = (uint_alias*)&AHs[buf][row * SA + ac];
      uint_alias* pL = (uint_alias*)&ALs[buf][row * SA + ac];
      pH[0] = pack2(h[0], h[1]); pH[1] = pack2(h[2], h[3]);
      pL[0] = pack2(l[0], l[1]); pL[1] = pack2(l[2], l[3]);
    }
    // B: pack (k,k+1) pairs into dwords, transposed store [n][k]
#pragma unroll
    for (int i = 0; i < 2; ++i) {
      const float* e0 = (const float*)&bReg[2 * i];      // k = 2*bkp + 16i
      const float* e1 = (const float*)&bReg[2 * i + 1];  // k = 2*bkp + 16i + 1
      const int kofs = 2 * bkp + 16 * i;
#pragma unroll
      for (int j = 0; j < 4; ++j) {
        __bf16 h0, l0, h1, l1;
        bsplit(e0[j], h0, l0);
        bsplit(e1[j], h1, l1);
        *(uint_alias*)&BHs[buf][(bn4 + j) * SA + kofs] = pack2(h0, h1);
        *(uint_alias*)&BLs[buf][(bn4 + j) * SA + kofs] = pack2(l0, l1);
      }
    }
  };

  // ---- wave tiling: 8 waves in 4(M) x 2(N), each wave owns 32x64 ----
  const int lane  = t & 31, wid = t >> 5;
  const int waveM = wid & 3, waveN = wid >> 2;
  const int half  = lane >> 4, lm = lane & 15;

  v8f acc[2][4] = {};

  loadRegs(0);
  stageLDS(0);
  __syncthreads();

  for (int kb = 0; kb < KBLOCKS; ++kb) {
    const int cur = kb & 1;
    const bool more = (kb + 1 < KBLOCKS);
    if (more) loadRegs(kb + 1);          // global prefetch, lands during wmma

    v16bf aH[2], aL[2], bH[4], bL[4];
#pragma unroll
    for (int mi = 0; mi < 2; ++mi) {
      const int mRow = waveM * 32 + mi * 16 + lm;
      aH[mi] = load_frag(&AHs[cur][mRow * SA], half);
      aL[mi] = load_frag(&ALs[cur][mRow * SA], half);
    }
#pragma unroll
    for (int ni = 0; ni < 4; ++ni) {
      const int nCol = waveN * 64 + ni * 16 + lm;
      bH[ni] = load_frag(&BHs[cur][nCol * SA], half);
      bL[ni] = load_frag(&BLs[cur][nCol * SA], half);
    }

#pragma unroll
    for (int mi = 0; mi < 2; ++mi)
#pragma unroll
      for (int ni = 0; ni < 4; ++ni) {
        v8f c = acc[mi][ni];
        // bf16x3 fp32 emulation: hi*hi + hi*lo + lo*hi
        c = __builtin_amdgcn_wmma_f32_16x16x32_bf16(false, aH[mi], false, bH[ni], (short)0, c, false, false);
        c = __builtin_amdgcn_wmma_f32_16x16x32_bf16(false, aH[mi], false, bL[ni], (short)0, c, false, false);
        c = __builtin_amdgcn_wmma_f32_16x16x32_bf16(false, aL[mi], false, bH[ni], (short)0, c, false, false);
        acc[mi][ni] = c;
      }

    if (more) stageLDS(cur ^ 1);         // write other buffer while peers still read cur
    __syncthreads();
  }

  // ---- epilogue: C layout (VGPR r -> row r + 8*half, col lm), add bias ----
#pragma unroll
  for (int ni = 0; ni < 4; ++ni) {
    const int n = n0 + waveN * 64 + ni * 16 + lm;
    const float bv = bias[n];
#pragma unroll
    for (int mi = 0; mi < 2; ++mi) {
      const int mBase = m0 + waveM * 32 + mi * 16 + half * 8;
#pragma unroll
      for (int r = 0; r < 8; ++r)
        out[(size_t)(mBase + r) * NOUT + n] = acc[mi][ni][r] + bv;
    }
  }
}

extern "C" void kernel_launch(void* const* d_in, const int* in_sizes, int n_in,
                              void* d_out, int out_size, void* d_ws, size_t ws_size,
                              hipStream_t stream) {
  (void)in_sizes; (void)n_in; (void)out_size; (void)d_ws; (void)ws_size;
  const float* X    = (const float*)d_in[0];   // [M, IN] fp32
  const float* W    = (const float*)d_in[1];   // [E, IN, OUT] fp32
  const int*   GS   = (const int*)d_in[2];     // [E] int32
  const float* bias = (const float*)d_in[3];   // [OUT] fp32
  float*       out  = (float*)d_out;           // [M, OUT] fp32

  dim3 grid(NOUT / BN, MTOK / BM);             // 64 x 64 tiles
  dim3 block(256);                             // 8 wave32 waves
  moe_grouped_gemm_bf16x3<<<grid, block, 0, stream>>>(X, W, GS, bias, out);
}
